// CenterModule_46574625357894
// MI455X (gfx1250) — matrix operand order repaired and checked by reading, not compiled, not run on hardware
//
#include <hip/hip_runtime.h>
#include <hip/hip_bf16.h>
#include <math.h>
#include <stdint.h>

typedef __attribute__((ext_vector_type(16))) _Float16 v16h;
typedef __attribute__((ext_vector_type(8)))  _Float16 v8h;
typedef __attribute__((ext_vector_type(4)))  _Float16 v4h;
typedef __attribute__((ext_vector_type(8)))  float    v8f;
typedef __attribute__((ext_vector_type(4)))  float    v4f;

#define N_ANCHORS    4096
#define N_CANDS      50000
#define N_CENTERS    10000
#define DIM          768
#define NEG          16
#define SUBT         5                       // A-subtiles per wave (hazard distance = 5)
#define STEP         (8 * SUBT * 16)         // 640 columns per block iteration

// padded row counts so tail loads need no clamping (base + imm offsets only)
#define N_CANDS_PAD    (((N_CANDS   + STEP - 1) / STEP) * STEP)   // 50560
#define N_CENTERS_PAD  (((N_CENTERS + STEP - 1) / STEP) * STEP)   // 10240

// ---------------- workspace layout ----------------
#define WS_FIXED_BYTES ((size_t)(N_CENTERS + N_ANCHORS) * 4 + (size_t)N_ANCHORS * NEG * 4)
#define H_CAND_P ((size_t)N_CANDS_PAD   * DIM)
#define H_CENT_P ((size_t)N_CENTERS_PAD * DIM)
#define H_ANCH   ((size_t)N_ANCHORS     * DIM)
#define WS_REQ_BYTES (WS_FIXED_BYTES + 2 * (H_CAND_P + H_CENT_P + H_ANCH) * sizeof(_Float16))

// ---------------------------------------------------------------------------
// Row squared-norms for centers and anchors (one block per row).
// ---------------------------------------------------------------------------
__global__ __launch_bounds__(256)
void row_norms_kernel(const float* __restrict__ centers,
                      const float* __restrict__ anchors,
                      float* __restrict__ c2, float* __restrict__ a2) {
  __shared__ float red[256];
  const int b = blockIdx.x;
  const float* row;
  float* outp;
  if (b < N_CENTERS) { row = centers + (size_t)b * DIM;               outp = c2 + b; }
  else               { row = anchors + (size_t)(b - N_CENTERS) * DIM; outp = a2 + (b - N_CENTERS); }
  float s = 0.f;
  for (int k = threadIdx.x; k < DIM; k += 256) { float v = row[k]; s += v * v; }
  red[threadIdx.x] = s;
  __syncthreads();
  for (int off = 128; off > 0; off >>= 1) {
    if (threadIdx.x < off) red[threadIdx.x] += red[threadIdx.x + off];
    __syncthreads();
  }
  if (threadIdx.x == 0) *outp = red[0];
}

// ---------------------------------------------------------------------------
// Split f32 rows into hi/lo f16 (row-major, vectorized float4 -> 2x v4h).
// ---------------------------------------------------------------------------
__global__ __launch_bounds__(256)
void split_kernel(const float* __restrict__ x, _Float16* __restrict__ hi,
                  _Float16* __restrict__ lo, int n4) {
  const int i = blockIdx.x * 256 + threadIdx.x;
  if (i >= n4) return;
  v4f v = ((const v4f*)x)[i];
  v4h h, l;
#pragma unroll
  for (int t = 0; t < 4; ++t) {
    _Float16 hh = (_Float16)v[t];
    h[t] = hh;
    l[t] = (_Float16)(v[t] - (float)hh);
  }
  ((v4h*)hi)[i] = h;
  ((v4h*)lo)[i] = l;
}

// ---------------------------------------------------------------------------
// Split + transpose anchors: [4096][768] f32 -> [768][4096] hi/lo f16.
// ---------------------------------------------------------------------------
__global__ __launch_bounds__(256)
void transpose_split_anchors_kernel(const float* __restrict__ anchors,
                                    _Float16* __restrict__ thi,
                                    _Float16* __restrict__ tlo) {
  const int m = blockIdx.x;
  for (int k = threadIdx.x; k < DIM; k += 256) {
    float v = anchors[(size_t)m * DIM + k];
    _Float16 h = (_Float16)v;
    thi[(size_t)k * N_ANCHORS + m] = h;
    tlo[(size_t)k * N_ANCHORS + m] = (_Float16)(v - (float)h);
  }
}

// ---------------------------------------------------------------------------
// PRE path: fused split-f16 WMMA GEMM + streaming reduction.
// Block = 256 thr (8 waves) owns 16 anchors.  Each wave: 5 A-subtiles (80
// rows, padded arrays -> no clamps) vs one register B fragment: 15 WMMAs per
// B load, accumulators interleaved 5 apart to cover the f16 WMMA hazard.
// Whole K loop unrolled: every fragment load is ONE base + imm offset.
// B tile staged via global_load_async_to_lds_b128 (ASYNCcnt DMA path).
// ---------------------------------------------------------------------------
template <int MODE>
__global__ __launch_bounds__(256)
void simtop_pre_kernel(const _Float16* __restrict__ anchT_hi,   // [768][4096]
                       const _Float16* __restrict__ anchT_lo,
                       const _Float16* __restrict__ m_hi,       // [nmat_pad][768]
                       const _Float16* __restrict__ m_lo,
                       int nmat,
                       const float* __restrict__ c2,
                       const float* __restrict__ a2,
                       int*   __restrict__ hard_idx,
                       float* __restrict__ assign_out,
                       float* __restrict__ dist_out) {
  __shared__ __align__(16) _Float16 sBhi[DIM * 16];     // 24 KB
  __shared__ __align__(16) _Float16 sBlo[DIM * 16];     // 24 KB
  __shared__ __align__(16) float    sSims[8 * 16 * 16]; //  8 KB

  const int tid  = threadIdx.x;
  const int wave = tid >> 5;
  const int lane = tid & 31;
  const int g    = lane >> 4;
  const int mloc = lane & 15;
  const int row0 = blockIdx.x * 16;

  // ---- async-DMA stage of the 16-anchor B tile into LDS ----
  for (int o = tid * 8; o < DIM * 16; o += 256 * 8) {
    const int k = o >> 4, m = o & 15;                  // m in {0,8}
    const _Float16* ghi = &anchT_hi[(size_t)k * N_ANCHORS + row0 + m];
    const _Float16* glo = &anchT_lo[(size_t)k * N_ANCHORS + row0 + m];
    unsigned dhi = (unsigned)(uintptr_t)&sBhi[o];      // LDS byte address
    unsigned dlo = (unsigned)(uintptr_t)&sBlo[o];
    asm volatile("global_load_async_to_lds_b128 %0, %1, off"
                 :: "v"(dhi), "v"(ghi) : "memory");
    asm volatile("global_load_async_to_lds_b128 %0, %1, off"
                 :: "v"(dlo), "v"(glo) : "memory");
  }
  asm volatile("s_wait_asynccnt 0x0" ::: "memory");
  __syncthreads();

  float topv[NEG];
  int   topi[NEG];
  float bestv = -3.4e38f;
  int   besti = 0;
#pragma unroll
  for (int j = 0; j < NEG; ++j) { topv[j] = -3.4e38f; topi[j] = 0; }

  const int nIter = (nmat + STEP - 1) / STEP;          // arrays padded to nIter*STEP rows
  for (int it = 0; it < nIter; ++it) {
    const int nb = it * STEP;
    const int cb = nb + wave * (SUBT * 16);

    // ONE base pointer pair per iteration; subtiles/k-blocks are imm offsets
    const size_t base = (size_t)(cb + mloc) * DIM + g * 8;
    const _Float16* pah = m_hi + base;
    const _Float16* pal = m_lo + base;

    if (cb + STEP + mloc < nmat)
      __builtin_prefetch(m_hi + (size_t)(cb + STEP + mloc) * DIM, 0, 1);

    v8f c[SUBT];
#pragma unroll
    for (int j = 0; j < SUBT; ++j) c[j] = {};

#pragma unroll
    for (int kb = 0; kb < DIM; kb += 32) {
      // B fragment: lane holds anchor col mloc, K rows kb+8g+t and kb+16+8g+t
      v16h bh, bl;
#pragma unroll
      for (int t = 0; t < 8; ++t) {
        const int k0 = (kb + g * 8 + t) * 16 + mloc, k1 = k0 + 16 * 16;
        bh[t] = sBhi[k0]; bh[t + 8] = sBhi[k1];
        bl[t] = sBlo[k0]; bl[t + 8] = sBlo[k1];
      }

      // A fragments for all 5 subtiles: base + (j*16*DIM + kb [+16]) imm
      v16h AH[SUBT], AL[SUBT];
#pragma unroll
      for (int j = 0; j < SUBT; ++j) {
        const int off = j * 16 * DIM + kb;
        v8h a0 = *(const v8h*)(pah + off);
        v8h a1 = *(const v8h*)(pah + off + 16);
        v8h l0 = *(const v8h*)(pal + off);
        v8h l1 = *(const v8h*)(pal + off + 16);
#pragma unroll
        for (int t = 0; t < 8; ++t) {
          AH[j][t] = a0[t]; AH[j][t + 8] = a1[t];
          AL[j][t] = l0[t]; AL[j][t + 8] = l1[t];
        }
      }

      // interleaved accumulators: dependent WMMAs are SUBT=5 slots apart
#pragma unroll
      for (int j = 0; j < SUBT; ++j)
        c[j] = __builtin_amdgcn_wmma_f32_16x16x32_f16(false, AH[j], false, bh, (short)0, c[j], false, false);
#pragma unroll
      for (int j = 0; j < SUBT; ++j)
        c[j] = __builtin_amdgcn_wmma_f32_16x16x32_f16(false, AH[j], false, bl, (short)0, c[j], false, false);
#pragma unroll
      for (int j = 0; j < SUBT; ++j)
        c[j] = __builtin_amdgcn_wmma_f32_16x16x32_f16(false, AL[j], false, bh, (short)0, c[j], false, false);
    }

    // ---- spill + scan one 16-candidate subtile at a time (sims = 8 KB) ----
#pragma unroll
    for (int j = 0; j < SUBT; ++j) {
#pragma unroll
      for (int r = 0; r < 8; ++r)
        sSims[wave * 256 + (r + 8 * g) * 16 + mloc] = c[j][r];
      __syncthreads();

      if (tid < 16) {
        const int n    = tid;
        const int grow = row0 + n;
        for (int w = 0; w < 8; ++w) {
#pragma unroll
          for (int m = 0; m < 16; ++m) {
            const int cidx = nb + w * (SUBT * 16) + j * 16 + m;
            const float v  = sSims[w * 256 + m * 16 + n];
            if (cidx < nmat) {
              if (MODE == 0) {
                const float metric = v - 0.5f * c2[cidx];
                if (metric > bestv) { bestv = metric; besti = cidx; }
              } else {
                if (cidx != grow && v > topv[0]) {
                  topv[0] = v; topi[0] = cidx;
#pragma unroll
                  for (int q = 0; q < NEG - 1; ++q) {
                    if (topv[q] > topv[q + 1]) {
                      float tv = topv[q]; topv[q] = topv[q + 1]; topv[q + 1] = tv;
                      int   ti = topi[q]; topi[q] = topi[q + 1]; topi[q + 1] = ti;
                    }
                  }
                }
              }
            }
          }
        }
      }
      __syncthreads();
    }
  }

  if (tid < 16) {
    const int grow = row0 + tid;
    if (MODE == 0) {
      const float d2 = a2[grow] - 2.0f * bestv;
      assign_out[grow] = (float)besti;
      dist_out[grow]   = sqrtf(fmaxf(d2, 0.0f));
    } else {
#pragma unroll
      for (int j = 0; j < NEG; ++j)
        hard_idx[grow * NEG + j] = topi[NEG - 1 - j];
    }
  }
}

// ---------------------------------------------------------------------------
// Fallback path (small workspace): converts f32->hi/lo f16 inline.
// ---------------------------------------------------------------------------
template <int MODE>
__global__ __launch_bounds__(256)
void simtop_inline_kernel(const float* __restrict__ anchors,
                          const float* __restrict__ mat, int nmat,
                          const float* __restrict__ c2,
                          const float* __restrict__ a2,
                          int*   __restrict__ hard_idx,
                          float* __restrict__ assign_out,
                          float* __restrict__ dist_out) {
  __shared__ __align__(16) _Float16 sBhi[DIM * 16];
  __shared__ __align__(16) _Float16 sBlo[DIM * 16];
  __shared__ __align__(16) float    sSims[8 * 16 * 16];

  const int tid  = threadIdx.x;
  const int wave = tid >> 5;
  const int lane = tid & 31;
  const int g    = lane >> 4;
  const int mloc = lane & 15;
  const int row0 = blockIdx.x * 16;

  for (int idx = tid; idx < DIM * 16; idx += 256) {
    int m = idx / DIM;
    int k = idx - m * DIM;
    float v = anchors[(size_t)(row0 + m) * DIM + k];
    _Float16 h = (_Float16)v;
    sBhi[k * 16 + m] = h;
    sBlo[k * 16 + m] = (_Float16)(v - (float)h);
  }
  __syncthreads();

  float topv[NEG];
  int   topi[NEG];
  float bestv = -3.4e38f;
  int   besti = 0;
#pragma unroll
  for (int j = 0; j < NEG; ++j) { topv[j] = -3.4e38f; topi[j] = 0; }

  const int nIter = (nmat + 127) / 128;
  for (int it = 0; it < nIter; ++it) {
    const int nb = it * 128;
    const int cb = nb + wave * 16;
    const int myrow  = cb + mloc;
    const int rclamp = (myrow < nmat) ? myrow : 0;
    const float* arow = mat + (size_t)rclamp * DIM;

    if (cb + 128 + mloc < nmat)
      __builtin_prefetch(mat + (size_t)(cb + 128 + mloc) * DIM, 0, 1);

    v8f c = {};

    for (int kb = 0; kb < DIM; kb += 32) {
      const int kb0 = kb + g * 8;
      v4f f0 = *(const v4f*)(arow + kb0);
      v4f f1 = *(const v4f*)(arow + kb0 + 4);
      v4f f2 = *(const v4f*)(arow + kb0 + 16);
      v4f f3 = *(const v4f*)(arow + kb0 + 20);

      v16h ah, al;
#pragma unroll
      for (int t = 0; t < 16; ++t) {
        float v = (t < 4) ? f0[t] : (t < 8) ? f1[t - 4] : (t < 12) ? f2[t - 8] : f3[t - 12];
        _Float16 h = (_Float16)v;
        ah[t] = h;
        al[t] = (_Float16)(v - (float)h);
      }

      v16h bh, bl;
#pragma unroll
      for (int t = 0; t < 8; ++t) {
        const int k0 = (kb + g * 8 + t) * 16 + mloc, k1 = k0 + 16 * 16;
        bh[t] = sBhi[k0]; bh[t + 8] = sBhi[k1];
        bl[t] = sBlo[k0]; bl[t + 8] = sBlo[k1];
      }

      c = __builtin_amdgcn_wmma_f32_16x16x32_f16(false, ah, false, bh, (short)0, c, false, false);
      c = __builtin_amdgcn_wmma_f32_16x16x32_f16(false, ah, false, bl, (short)0, c, false, false);
      c = __builtin_amdgcn_wmma_f32_16x16x32_f16(false, al, false, bh, (short)0, c, false, false);
    }

#pragma unroll
    for (int r = 0; r < 8; ++r)
      sSims[wave * 256 + (r + 8 * g) * 16 + mloc] = c[r];
    __syncthreads();

    if (tid < 16) {
      const int n    = tid;
      const int grow = row0 + n;
      for (int w = 0; w < 8; ++w) {
#pragma unroll
        for (int m = 0; m < 16; ++m) {
          const int cidx = nb + w * 16 + m;
          const float v  = sSims[w * 256 + m * 16 + n];
          if (cidx < nmat) {
            if (MODE == 0) {
              const float metric = v - 0.5f * c2[cidx];
              if (metric > bestv) { bestv = metric; besti = cidx; }
            } else {
              if (cidx != grow && v > topv[0]) {
                topv[0] = v; topi[0] = cidx;
#pragma unroll
                for (int q = 0; q < NEG - 1; ++q) {
                  if (topv[q] > topv[q + 1]) {
                    float tv = topv[q]; topv[q] = topv[q + 1]; topv[q + 1] = tv;
                    int   ti = topi[q]; topi[q] = topi[q + 1]; topi[q + 1] = ti;
                  }
                }
              }
            }
          }
        }
      }
    }
    __syncthreads();
  }

  if (tid < 16) {
    const int grow = row0 + tid;
    if (MODE == 0) {
      const float d2 = a2[grow] - 2.0f * bestv;
      assign_out[grow] = (float)besti;
      dist_out[grow]   = sqrtf(fmaxf(d2, 0.0f));
    } else {
#pragma unroll
      for (int j = 0; j < NEG; ++j)
        hard_idx[grow * NEG + j] = topi[NEG - 1 - j];
    }
  }
}

// ---------------------------------------------------------------------------
// Gather hard negatives: one block per (anchor, k) pair, float4 streaming.
// ---------------------------------------------------------------------------
__global__ __launch_bounds__(192)
void gather_kernel(const float* __restrict__ cand,
                   const int* __restrict__ hard_idx,
                   float* __restrict__ out) {
  const int pair = blockIdx.x;
  const int idx  = hard_idx[pair];
  const v4f* src = (const v4f*)(cand + (size_t)idx * DIM);
  v4f*       dst = (v4f*)(out + (size_t)pair * DIM);
  dst[threadIdx.x] = src[threadIdx.x];
}

// ---------------------------------------------------------------------------
extern "C" void kernel_launch(void* const* d_in, const int* in_sizes, int n_in,
                              void* d_out, int out_size, void* d_ws, size_t ws_size,
                              hipStream_t stream) {
  const float* anchors    = (const float*)d_in[0];
  const float* candidates = (const float*)d_in[1];
  const float* centers    = (const float*)d_in[2];

  float* out        = (float*)d_out;
  float* hn         = out;
  float* assign_out = out + (size_t)N_ANCHORS * NEG * DIM;
  float* dist_out   = assign_out + N_ANCHORS;

  float* ws       = (float*)d_ws;
  float* c2       = ws;
  float* a2       = ws + N_CENTERS;
  int*   hard_idx = (int*)(ws + N_CENTERS + N_ANCHORS);

  row_norms_kernel<<<N_CENTERS + N_ANCHORS, 256, 0, stream>>>(centers, anchors, c2, a2);

  if (ws_size >= WS_REQ_BYTES) {
    _Float16* hbase    = (_Float16*)((char*)d_ws + WS_FIXED_BYTES);
    _Float16* cand_hi  = hbase;                      // padded to N_CANDS_PAD rows
    _Float16* cand_lo  = cand_hi + H_CAND_P;
    _Float16* cent_hi  = cand_lo + H_CAND_P;         // padded to N_CENTERS_PAD rows
    _Float16* cent_lo  = cent_hi + H_CENT_P;
    _Float16* anchT_hi = cent_lo + H_CENT_P;
    _Float16* anchT_lo = anchT_hi + H_ANCH;

    const int cand_n4 = (int)((size_t)N_CANDS   * DIM / 4);
    const int cent_n4 = (int)((size_t)N_CENTERS * DIM / 4);
    split_kernel<<<(cand_n4 + 255) / 256, 256, 0, stream>>>(candidates, cand_hi, cand_lo, cand_n4);
    split_kernel<<<(cent_n4 + 255) / 256, 256, 0, stream>>>(centers, cent_hi, cent_lo, cent_n4);
    transpose_split_anchors_kernel<<<N_ANCHORS, 256, 0, stream>>>(anchors, anchT_hi, anchT_lo);

    simtop_pre_kernel<0><<<N_ANCHORS / 16, 256, 0, stream>>>(
        anchT_hi, anchT_lo, cent_hi, cent_lo, N_CENTERS, c2, a2,
        nullptr, assign_out, dist_out);
    simtop_pre_kernel<1><<<N_ANCHORS / 16, 256, 0, stream>>>(
        anchT_hi, anchT_lo, cand_hi, cand_lo, N_CANDS, nullptr, nullptr,
        hard_idx, nullptr, nullptr);
  } else {
    simtop_inline_kernel<0><<<N_ANCHORS / 16, 256, 0, stream>>>(
        anchors, centers, N_CENTERS, c2, a2, nullptr, assign_out, dist_out);
    simtop_inline_kernel<1><<<N_ANCHORS / 16, 256, 0, stream>>>(
        anchors, candidates, N_CANDS, nullptr, nullptr, hard_idx, nullptr, nullptr);
  }

  gather_kernel<<<N_ANCHORS * NEG, 192, 0, stream>>>(candidates, hard_idx, hn);
}
